// SubgraphCompletionDecoder_19645180412343
// MI455X (gfx1250) — compile-verified, matrix operand order checked by reading, not run on hardware
//
#include <hip/hip_runtime.h>

// ---------------------------------------------------------------------------
// CDNA5 (gfx1250, wave32) implementation of SubgraphCompletionDecoder forward.
// Heavy GEMMs (memory K/V projections, symbol-side bilinear logits) use
// v_wmma_f32_16x16x32_f16; cross-attention is fused flash-style streaming over
// the ragged node lists with ASYNC global->LDS double buffering (ASYNCcnt).
// ---------------------------------------------------------------------------

typedef __attribute__((ext_vector_type(16))) _Float16 v16h;
typedef __attribute__((ext_vector_type(8)))  float    v8f;

#define DEV __device__ __forceinline__

// model dims
#define H_     128
#define NH_    4
#define HD_    32
#define KS     6
#define MA     4
#define BB     64
#define MAXSYM 1024
#define MAXALL 9216
#define SPLIT  8
#define TT     80
#define TN     32          // streaming tile rows (cross-attn)

// --- flattened param offsets (jax tree_flatten: dict keys sorted, recursive) ---
constexpr int P_ACTIVE_B = 0;
constexpr int P_ACTIVE_W = 1;
constexpr int P_ARG_FK_B = 129;
constexpr int P_ARG_FK_W = 257;
constexpr int P_ARG_FQ_B = 16641;
constexpr int P_ARG_FQ_W = 16769;
constexpr int P_ARG_SLOT = 33153;   // [6][4][128]
constexpr int P_ARG_TYPE_B = 36225;
constexpr int P_ARG_TYPE_W = 36228;
constexpr int P_ARG_VAR_B = 36612;
constexpr int P_ARG_VAR_W = 36632;
constexpr int P_CA1 = 39192;        // attn block: k.b,k.w,o.b,o.w,q.b,q.w,v.b,v.w
constexpr int P_CA2 = 105240;
constexpr int P_FFN1 = 171288;      // b1(512), b2(128), w1(65536), w2(65536)
constexpr int P_FFN2 = 303000;
constexpr int P_LIT_SLOT = 434712;  // [6][128]
constexpr int P_LN1 = 435480;       // b(128), g(128)
constexpr int P_LN2 = 435736;
constexpr int P_LN3 = 435992;
constexpr int P_LN4 = 436248;
constexpr int P_LN5 = 436504;
constexpr int P_LN6 = 436760;
constexpr int P_POL_B = 437016;
constexpr int P_POL_W = 437017;
constexpr int P_PRED_K_B = 437145;
constexpr int P_PRED_K_W = 437273;
constexpr int P_PRED_Q_B = 453657;
constexpr int P_PRED_Q_W = 453785;
constexpr int P_SA1 = 470169;
constexpr int P_SA2 = 536217;
// attn sub-offsets
constexpr int A_KB = 0,  A_KW = 128,   A_OB = 16512, A_OW = 16640;
constexpr int A_QB = 33024, A_QW = 33152, A_VB = 49536, A_VW = 49664;
// ffn sub-offsets
constexpr int F_B1 = 0, F_B2 = 512, F_W1 = 640, F_W2 = 66176;

// --- output offsets (tuple concatenated flat, floats) ---
constexpr int OFF_ACT = 0;            // [64,6]
constexpr int OFF_POL = 384;          // [64,6]
constexpr int OFF_PRED = 768;         // [64,6,1024]
constexpr int OFF_ATYPE = 393984;     // [64,6,4,3]
constexpr int OFF_AVAR = 398592;      // [64,6,4,20]
constexpr int OFF_AFUNC = 429312;     // [64,6,4,1024]
constexpr int OFF_TACT = 2002176;
constexpr int OFF_TPOL = 2002560;
constexpr int OFF_TPRED = 2002944;
constexpr int OFF_TATYPE = 2003328;
constexpr int OFF_TAVAR = 2004864;
constexpr int OFF_TAFUNC = 2006400;

// ---------------------------------------------------------------------------
// WMMA helpers.  A-matrix 16x32 f16 layout (ISA 7.12.2): lane&15 = row M,
// lane>>4 selects K halves; VGPR j holds K pair {base, base+1} with
// base = (j<4 ? 2j : 16+2(j-4)) + (lane>=16 ? 8 : 0).  B mirrors with lane = N.
// ---------------------------------------------------------------------------
DEV v16h frag_ld(const _Float16* base, int ld, int r0, int k0, int lane) {
  const _Float16* rp = base + (size_t)(r0 + (lane & 15)) * ld + k0 + ((lane >> 4) << 3);
  v16h f;
#pragma unroll
  for (int j = 0; j < 8; ++j) {
    int kb = (j < 4) ? (2 * j) : (16 + 2 * (j - 4));
    f[2 * j]     = rp[kb];
    f[2 * j + 1] = rp[kb + 1];
  }
  return f;
}

// fragment built from fp32 storage with inline f32->f16 convert
DEV v16h frag_ldf(const float* base, int ld, int r0, int k0, int lane) {
  const float* rp = base + (size_t)(r0 + (lane & 15)) * ld + k0 + ((lane >> 4) << 3);
  v16h f;
#pragma unroll
  for (int j = 0; j < 8; ++j) {
    int kb = (j < 4) ? (2 * j) : (16 + 2 * (j - 4));
    f[2 * j]     = (_Float16)rp[kb];
    f[2 * j + 1] = (_Float16)rp[kb + 1];
  }
  return f;
}

// Pre-swizzled weight fragment: layout [tn(8)][tk(4)][lane(32)][16 halves]
DEV v16h wfrag_ld(const _Float16* wf, int tk, int tn, int lane) {
  return *(const v16h*)(wf + ((((tn * 4 + tk) * 32) + lane) * 16));
}

DEV v8f wmma16(v16h a, v16h b, v8f c) {
  return __builtin_amdgcn_wmma_f32_16x16x32_f16(false, a, false, b, (short)0, c, false, false);
}

// Async copy of one TN x 128 fp32 tile into LDS (4 x b128 per wave, ASYNCcnt).
// Rows beyond row_lim are clamped (garbage-but-safe; masked later via P=0).
DEV void async_load_tile(const float* gbase, int row_lim, int tile,
                         unsigned lds_base, int wv, int lane) {
#pragma unroll
  for (int j = 0; j < 4; ++j) {
    int c = wv * 128 + j * 32 + lane;       // 16B chunk id, 1024 chunks = 16KB
    int r = c >> 5, col = c & 31;           // 32 chunks per 512B row
    int gr = tile * TN + r;
    if (gr > row_lim) gr = row_lim;
    const float* ga = gbase + (size_t)gr * H_ + col * 4;
    unsigned la = lds_base + (unsigned)c * 16u;
    asm volatile("global_load_async_to_lds_b128 %0, %1, off"
                 :: "v"(la), "v"(ga) : "memory");
  }
}

// ---------------------------------------------------------------------------
// Prep: per-sample offsets (batch arrays are sorted), weight swizzle -> f16
// fragments, slots0 = broadcast(lit_slot).
// ---------------------------------------------------------------------------
__global__ __launch_bounds__(256)
void k_prep(const float* __restrict__ prm,
            const int* __restrict__ sym_batch, int nsym,
            const int* __restrict__ all_batch, int nall,
            int* __restrict__ sym_off, int* __restrict__ all_off,
            _Float16* __restrict__ wf, float* __restrict__ slots0) {
  int gi = blockIdx.x * 256 + threadIdx.x;
  if (gi < 6 * 16384) {
    int mat = gi >> 14, rem = gi & 16383;
    int e = rem & 15, lane = (rem >> 4) & 31, tk = (rem >> 9) & 3, tn = rem >> 11;
    int vj = e >> 1, lo = e & 1;
    int k = tk * 32 + ((vj < 4) ? (2 * vj) : (16 + 2 * (vj - 4))) + ((lane >> 4) << 3) + lo;
    int n = tn * 16 + (lane & 15);
    int srcbase;
    switch (mat) {
      case 0: srcbase = P_CA1 + A_KW; break;
      case 1: srcbase = P_CA1 + A_VW; break;
      case 2: srcbase = P_CA2 + A_KW; break;
      case 3: srcbase = P_CA2 + A_VW; break;
      case 4: srcbase = P_PRED_K_W;   break;
      default: srcbase = P_ARG_FK_W;  break;
    }
    wf[gi] = (_Float16)prm[srcbase + k * H_ + n];
  } else if (gi < 6 * 16384 + BB * KS * H_) {
    int i = gi - 6 * 16384;
    slots0[i] = prm[P_LIT_SLOT + (i % (KS * H_))];
  } else if (gi < 6 * 16384 + BB * KS * H_ + 130) {
    int i = gi - (6 * 16384 + BB * KS * H_);
    if (i < 65) {
      int v = i, lo = 0, hi = nsym;
      while (lo < hi) { int mid = (lo + hi) >> 1; if (sym_batch[mid] < v) lo = mid + 1; else hi = mid; }
      sym_off[v] = lo;
    } else {
      int v = i - 65, lo = 0, hi = nall;
      while (lo < hi) { int mid = (lo + hi) >> 1; if (all_batch[mid] < v) lo = mid + 1; else hi = mid; }
      all_off[v] = lo;
    }
  }
}

// ---------------------------------------------------------------------------
// Fused streaming cross-attention (flash style, split-K over memory rows).
// Grid = B*SPLIT, block = 256 (8 wave32).  Per TN-row tile (double buffered
// via GLOBAL_LOAD_ASYNC_TO_LDS + s_wait_asynccnt):
//   K,V = X@W via WMMA (A-frags converted f32->f16 from the async stage);
//   scores via WMMA ; online softmax ; partial o/m/l to workspace.
// ---------------------------------------------------------------------------
__global__ __launch_bounds__(256)
void k_cross_attn(const float* __restrict__ all_x, const float* __restrict__ prm,
                  const int* __restrict__ all_off, const float* __restrict__ slots_in,
                  const _Float16* __restrict__ wfK, const _Float16* __restrict__ wfV,
                  int attn_base,
                  float* __restrict__ o_part, float* __restrict__ m_part,
                  float* __restrict__ l_part) {
  __shared__ __align__(16) float stage[2][TN * H_];   // 2 x 16KB async stage
  __shared__ _Float16 Kp[TN][H_];
  __shared__ _Float16 Vp[TN][H_];
  __shared__ _Float16 Ql[16][H_];
  __shared__ float Sl[NH_][KS][TN];
  __shared__ float Pl[NH_][KS][TN];
  __shared__ float mstate[24], lstate[24], sstate[24];
  __shared__ float bK[H_], bV[H_];

  const int t = threadIdx.x, lane = t & 31, wv = t >> 5;
  const int b = blockIdx.x >> 3, sp = blockIdx.x & 7;

  for (int i = t; i < H_; i += 256) { bK[i] = prm[attn_base + A_KB + i]; bV[i] = prm[attn_base + A_VB + i]; }
  for (int i = t; i < 16 * H_; i += 256) Ql[i >> 7][i & 127] = (_Float16)0.f;
  if (t < 24) { mstate[t] = -1e30f; lstate[t] = 0.f; sstate[t] = 1.f; }
  __syncthreads();
  // Q = slots @ Wq + bq (tiny; recomputed per split, negligible)
  const float* qw = prm + attn_base + A_QW;
  const float* qb = prm + attn_base + A_QB;
  const float* sin = slots_in + (size_t)b * KS * H_;
  for (int i = t; i < KS * H_; i += 256) {
    int q = i >> 7, o = i & 127;
    float acc = qb[o];
    for (int c = 0; c < H_; ++c) acc += sin[q * H_ + c] * qw[c * H_ + o];
    Ql[q][o] = (_Float16)acc;
  }

  int base = all_off[b];
  int cnt = all_off[b + 1] - base; if (cnt > MAXALL) cnt = MAXALL;
  int per = (cnt + SPLIT - 1) / SPLIT;
  int r0 = sp * per; if (r0 > cnt) r0 = cnt;
  int r1 = r0 + per; if (r1 > cnt) r1 = cnt;
  int nrows = r1 - r0;
  int ntiles = (nrows + TN - 1) / TN;

  const float* gxb = all_x + (size_t)(base + r0) * H_;
  const unsigned lds_stage0 = (unsigned)(size_t)&stage[0][0];
  const unsigned lds_stage1 = (unsigned)(size_t)&stage[1][0];

  float acc0 = 0.f, acc1 = 0.f, acc2 = 0.f, acc3 = 0.f;
  const int pp = t >> 3, sub = t & 7;                 // pair / hd-quad (t<192)
  const int ph = pp / KS, pq = pp % KS;

  if (ntiles > 0)
    async_load_tile(gxb, nrows - 1, 0, lds_stage0, wv, lane);

  for (int tile = 0; tile < ntiles; ++tile) {
    int nv = nrows - tile * TN; if (nv > TN) nv = TN;
    const float* Xs = &stage[tile & 1][0];
    if (tile + 1 < ntiles) {
      async_load_tile(gxb, nrows - 1, tile + 1,
                      ((tile + 1) & 1) ? lds_stage1 : lds_stage0, wv, lane);
      asm volatile("s_wait_asynccnt 0x4" ::: "memory");   // current tile landed
    } else {
      asm volatile("s_wait_asynccnt 0x0" ::: "memory");
    }
    __syncthreads();
    // K & V projections: 2x8 tiles x {K,V} = 32 output tiles, 4 per wave.
#pragma unroll
    for (int j = 0; j < 4; ++j) {
      int tid = wv * 4 + j;
      int isV = tid >> 4, tl = tid & 15;
      int tm = tl >> 3, tn = tl & 7;
      const _Float16* wfp = isV ? wfV : wfK;
      v8f c = {};
#pragma unroll
      for (int tk = 0; tk < 4; ++tk) {
        v16h a = frag_ldf(Xs, H_, tm * 16, tk * 32, lane);
        v16h bm = wfrag_ld(wfp, tk, tn, lane);
        c = wmma16(a, bm, c);
      }
      const float* bias = isV ? bV : bK;
      _Float16 (*Out)[H_] = isV ? Vp : Kp;
      int col = tn * 16 + (lane & 15);
      int rb = tm * 16 + ((lane >> 4) << 3);
#pragma unroll
      for (int i = 0; i < 8; ++i) Out[rb + i][col] = (_Float16)(c[i] + bias[col]);
    }
    __syncthreads();
    // scores: wave w = head w (K-dim = HD = 32 -> one WMMA per 16-row group)
    if (wv < 4) {
      int h = wv;
#pragma unroll
      for (int g = 0; g < 2; ++g) {
        v16h aq = frag_ld(&Ql[0][0], H_, 0, h * 32, lane);
        v16h bk = frag_ld(&Kp[0][0], H_, g * 16, h * 32, lane);
        v8f s = {};
        s = wmma16(aq, bk, s);
        if (lane < 16) {
#pragma unroll
          for (int i = 0; i < KS; ++i)
            Sl[h][i][g * 16 + lane] = s[i] * 0.17677669529663687f;  // 1/sqrt(32)
        }
      }
    }
    __syncthreads();
    if (t < 24) {                                      // online softmax stats
      int h = t / KS, q = t % KS;
      float mprev = mstate[t], tmax = -1e30f;
      for (int r = 0; r < nv; ++r) tmax = fmaxf(tmax, Sl[h][q][r]);
      float mnew = fmaxf(mprev, tmax);
      float sc = __expf(mprev - mnew);
      float ls = 0.f;
      for (int r = 0; r < TN; ++r) {
        float p = (r < nv) ? __expf(Sl[h][q][r] - mnew) : 0.f;
        Pl[h][q][r] = p; ls += p;
      }
      mstate[t] = mnew;
      lstate[t] = lstate[t] * sc + ls;
      sstate[t] = sc;
    }
    __syncthreads();
    if (t < 192) {                                     // acc += P @ V_head
      float sc = sstate[pp];
      acc0 *= sc; acc1 *= sc; acc2 *= sc; acc3 *= sc;
      const _Float16* vcol = &Vp[0][ph * 32 + sub * 4];
      const float* prow = &Pl[ph][pq][0];
      for (int r = 0; r < TN; ++r) {
        float p = prow[r];
        const _Float16* vr = vcol + r * H_;
        acc0 += p * (float)vr[0]; acc1 += p * (float)vr[1];
        acc2 += p * (float)vr[2]; acc3 += p * (float)vr[3];
      }
    }
    __syncthreads();
  }
  if (t < 192) {
    float* op = o_part + (((size_t)(b * SPLIT + sp) * 24 + pp) * 32 + sub * 4);
    op[0] = acc0; op[1] = acc1; op[2] = acc2; op[3] = acc3;
  }
  if (t < 24) {
    m_part[(b * SPLIT + sp) * 24 + t] = mstate[t];
    l_part[(b * SPLIT + sp) * 24 + t] = lstate[t];
  }
}

// ---------------------------------------------------------------------------
// Combine split partials, o-projection, residual, layernorm.  Grid=B.
// ---------------------------------------------------------------------------
__global__ __launch_bounds__(128)
void k_attn_combine(const float* __restrict__ prm, const float* __restrict__ slots_in,
                    const float* __restrict__ o_part, const float* __restrict__ m_part,
                    const float* __restrict__ l_part, int attn_base, int ln_base,
                    float* __restrict__ slots_out) {
  __shared__ float attn[KS][H_], pre[KS][H_];
  __shared__ float wcomb[24][SPLIT], linv[24];
  __shared__ float mu[KS], rs[KS];
  int t = threadIdx.x, b = blockIdx.x;
  if (t < 24) {
    float M = -1e30f;
    for (int s = 0; s < SPLIT; ++s) M = fmaxf(M, m_part[(b * SPLIT + s) * 24 + t]);
    float L = 0.f;
    for (int s = 0; s < SPLIT; ++s) {
      float w = __expf(m_part[(b * SPLIT + s) * 24 + t] - M);
      wcomb[t][s] = w;
      L += w * l_part[(b * SPLIT + s) * 24 + t];
    }
    linv[t] = 1.f / fmaxf(L, 1e-30f);
  }
  __syncthreads();
  for (int i = t; i < 24 * 32; i += 128) {
    int p = i >> 5, hd = i & 31, h = p / KS, q = p % KS;
    float o = 0.f;
    for (int s = 0; s < SPLIT; ++s)
      o += wcomb[p][s] * o_part[((size_t)(b * SPLIT + s) * 24 + p) * 32 + hd];
    attn[q][h * 32 + hd] = o * linv[p];
  }
  __syncthreads();
  const float* ow = prm + attn_base + A_OW;
  const float* ob = prm + attn_base + A_OB;
  const float* sin = slots_in + (size_t)b * KS * H_;
  for (int i = t; i < KS * H_; i += 128) {
    int q = i >> 7, o = i & 127;
    float acc = ob[o];
    for (int c = 0; c < H_; ++c) acc += attn[q][c] * ow[c * H_ + o];
    pre[q][o] = acc + sin[i];
  }
  __syncthreads();
  if (t < KS) {
    float m = 0.f; for (int c = 0; c < H_; ++c) m += pre[t][c];
    m *= (1.f / H_);
    float v = 0.f; for (int c = 0; c < H_; ++c) { float d = pre[t][c] - m; v += d * d; }
    mu[t] = m; rs[t] = rsqrtf(v * (1.f / H_) + 1e-5f);
  }
  __syncthreads();
  const float* lb = prm + ln_base;
  const float* lg = prm + ln_base + 128;
  float* so = slots_out + (size_t)b * KS * H_;
  for (int i = t; i < KS * H_; i += 128) {
    int q = i >> 7, c = i & 127;
    so[i] = (pre[q][c] - mu[q]) * rs[q] * lg[c] + lb[c];
  }
}

// ---------------------------------------------------------------------------
// Middle block: self-attn(6 tokens) + LN + FFN(128->512->128) + LN. Grid=B.
// ---------------------------------------------------------------------------
__global__ __launch_bounds__(128)
void k_mid(const float* __restrict__ prm, const float* __restrict__ slots_in,
           int sa_base, int lnA_base, int ffn_base, int lnB_base,
           float* __restrict__ slots_out) {
  __shared__ float x[KS][H_], qv[KS][H_], kv[KS][H_], vv[KS][H_];
  __shared__ float S[NH_][KS][KS];
  __shared__ float att[KS][H_], pre[KS][H_], y[KS][H_];
  __shared__ float h1[KS][4 * H_];
  __shared__ float mu[KS], rs[KS];
  int t = threadIdx.x, b = blockIdx.x;
  const float* sin = slots_in + (size_t)b * KS * H_;
  for (int i = t; i < KS * H_; i += 128) x[i >> 7][i & 127] = sin[i];
  __syncthreads();
  for (int i = t; i < 3 * KS * H_; i += 128) {
    int which = i / (KS * H_), r = i % (KS * H_);
    int q = r >> 7, o = r & 127;
    const float* w  = prm + sa_base + (which == 0 ? A_QW : which == 1 ? A_KW : A_VW);
    const float* bi = prm + sa_base + (which == 0 ? A_QB : which == 1 ? A_KB : A_VB);
    float acc = bi[o];
    for (int c = 0; c < H_; ++c) acc += x[q][c] * w[c * H_ + o];
    if (which == 0) qv[q][o] = acc; else if (which == 1) kv[q][o] = acc; else vv[q][o] = acc;
  }
  __syncthreads();
  if (t < NH_ * KS * KS) {
    int h = t / (KS * KS), r = t % (KS * KS), i = r / KS, j = r % KS;
    float s = 0.f;
    for (int d = 0; d < HD_; ++d) s += qv[i][h * HD_ + d] * kv[j][h * HD_ + d];
    S[h][i][j] = s * 0.17677669529663687f;
  }
  __syncthreads();
  if (t < NH_ * KS) {
    int h = t / KS, i = t % KS;
    float m = -1e30f;
    for (int j = 0; j < KS; ++j) m = fmaxf(m, S[h][i][j]);
    float e[KS], l = 0.f;
    for (int j = 0; j < KS; ++j) { e[j] = __expf(S[h][i][j] - m); l += e[j]; }
    float inv = 1.f / l;
    for (int j = 0; j < KS; ++j) S[h][i][j] = e[j] * inv;
  }
  __syncthreads();
  for (int i = t; i < KS * H_; i += 128) {
    int q = i >> 7, c = i & 127, h = c >> 5;
    float o = 0.f;
    for (int j = 0; j < KS; ++j) o += S[h][q][j] * vv[j][c];
    att[q][c] = o;
  }
  __syncthreads();
  for (int i = t; i < KS * H_; i += 128) {
    int q = i >> 7, o = i & 127;
    const float* ow = prm + sa_base + A_OW;
    float acc = prm[sa_base + A_OB + o];
    for (int c = 0; c < H_; ++c) acc += att[q][c] * ow[c * H_ + o];
    pre[q][o] = acc + x[q][o];
  }
  __syncthreads();
  if (t < KS) {
    float m = 0.f; for (int c = 0; c < H_; ++c) m += pre[t][c];
    m *= (1.f / H_);
    float v = 0.f; for (int c = 0; c < H_; ++c) { float d = pre[t][c] - m; v += d * d; }
    mu[t] = m; rs[t] = rsqrtf(v * (1.f / H_) + 1e-5f);
  }
  __syncthreads();
  for (int i = t; i < KS * H_; i += 128) {
    int q = i >> 7, c = i & 127;
    y[q][c] = (pre[q][c] - mu[q]) * rs[q] * prm[lnA_base + 128 + c] + prm[lnA_base + c];
  }
  __syncthreads();
  for (int i = t; i < KS * 4 * H_; i += 128) {
    int q = i / (4 * H_), o = i % (4 * H_);
    const float* w1 = prm + ffn_base + F_W1;
    float acc = prm[ffn_base + F_B1 + o];
    for (int c = 0; c < H_; ++c) acc += y[q][c] * w1[c * 4 * H_ + o];
    h1[q][o] = fmaxf(acc, 0.f);
  }
  __syncthreads();
  for (int i = t; i < KS * H_; i += 128) {
    int q = i >> 7, o = i & 127;
    const float* w2 = prm + ffn_base + F_W2;
    float acc = prm[ffn_base + F_B2 + o];
    for (int c = 0; c < 4 * H_; ++c) acc += h1[q][c] * w2[c * H_ + o];
    pre[q][o] = acc + y[q][o];
  }
  __syncthreads();
  if (t < KS) {
    float m = 0.f; for (int c = 0; c < H_; ++c) m += pre[t][c];
    m *= (1.f / H_);
    float v = 0.f; for (int c = 0; c < H_; ++c) { float d = pre[t][c] - m; v += d * d; }
    mu[t] = m; rs[t] = rsqrtf(v * (1.f / H_) + 1e-5f);
  }
  __syncthreads();
  float* so = slots_out + (size_t)b * KS * H_;
  for (int i = t; i < KS * H_; i += 128) {
    int q = i >> 7, c = i & 127;
    so[i] = (pre[q][c] - mu[q]) * rs[q] * prm[lnB_base + 128 + c] + prm[lnB_base + c];
  }
}

// ---------------------------------------------------------------------------
// Small heads on final slots: active/polarity, arg_type/arg_var logits;
// pred_q and arg-func queries written as zero-padded f16 A-tiles. Grid=B.
// ---------------------------------------------------------------------------
__global__ __launch_bounds__(128)
void k_heads(const float* __restrict__ prm, const float* __restrict__ slots,
             float* __restrict__ out, _Float16* __restrict__ predq_a,
             _Float16* __restrict__ afq_a) {
  __shared__ float x[KS][H_];
  __shared__ float as_[24][H_];
  int t = threadIdx.x, b = blockIdx.x;
  const float* sin = slots + (size_t)b * KS * H_;
  for (int i = t; i < KS * H_; i += 128) x[i >> 7][i & 127] = sin[i];
  _Float16* pq = predq_a + (size_t)b * 16 * H_;
  _Float16* fq = afq_a + (size_t)b * 32 * H_;
  for (int i = t; i < 16 * H_; i += 128) pq[i] = (_Float16)0.f;
  for (int i = t; i < 32 * H_; i += 128) fq[i] = (_Float16)0.f;
  __syncthreads();
  if (t < 2 * KS) {
    int which = t / KS, q = t % KS;
    const float* w = prm + (which ? P_POL_W : P_ACTIVE_W);
    float acc = prm[which ? P_POL_B : P_ACTIVE_B];
    for (int c = 0; c < H_; ++c) acc += x[q][c] * w[c];
    out[(which ? OFF_POL : OFF_ACT) + b * KS + q] = acc;
  }
  for (int i = t; i < KS * H_; i += 128) {
    int q = i >> 7, o = i & 127;
    float acc = prm[P_PRED_Q_B + o];
    for (int c = 0; c < H_; ++c) acc += x[q][c] * prm[P_PRED_Q_W + c * H_ + o];
    pq[q * H_ + o] = (_Float16)acc;
  }
  for (int i = t; i < 24 * H_; i += 128) {
    int r = i >> 7, c = i & 127, k = r >> 2;
    as_[r][c] = prm[P_ARG_SLOT + i] + x[k][c];
  }
  __syncthreads();
  for (int i = t; i < 24 * 3; i += 128) {
    int r = i / 3, c = i % 3;
    float acc = prm[P_ARG_TYPE_B + c];
    for (int d = 0; d < H_; ++d) acc += as_[r][d] * prm[P_ARG_TYPE_W + d * 3 + c];
    out[OFF_ATYPE + (b * 24 + r) * 3 + c] = acc;
  }
  for (int i = t; i < 24 * 20; i += 128) {
    int r = i / 20, c = i % 20;
    float acc = prm[P_ARG_VAR_B + c];
    for (int d = 0; d < H_; ++d) acc += as_[r][d] * prm[P_ARG_VAR_W + d * 20 + c];
    out[OFF_AVAR + (b * 24 + r) * 20 + c] = acc;
  }
  for (int i = t; i < 24 * H_; i += 128) {
    int r = i >> 7, o = i & 127;
    float acc = prm[P_ARG_FQ_B + o];
    for (int c = 0; c < H_; ++c) acc += as_[r][c] * prm[P_ARG_FQ_W + c * H_ + o];
    fq[r * H_ + o] = (_Float16)acc;
  }
}

// ---------------------------------------------------------------------------
// Symbol-side bilinear logits: project sym tile through pred_k / arg_fk (WMMA),
// then q @ k^T (WMMA, K=128 in 4 steps).  Padded rows = bias (matches the
// unmasked reference).  Grid = B * 16 column tiles, block = 256.
// ---------------------------------------------------------------------------
__global__ __launch_bounds__(256)
void k_sym_logits(const float* __restrict__ symbol_x, const float* __restrict__ prm,
                  const int* __restrict__ sym_off,
                  const _Float16* __restrict__ wfPK, const _Float16* __restrict__ wfFK,
                  const _Float16* __restrict__ predq_a, const _Float16* __restrict__ afq_a,
                  float* __restrict__ out) {
  __shared__ _Float16 Xl[64][H_];
  __shared__ _Float16 Kp[64][H_];
  __shared__ _Float16 Fk[64][H_];
  __shared__ float bPK[H_], bFK[H_];
  int t = threadIdx.x, lane = t & 31, wv = t >> 5;
  int b = blockIdx.x >> 4, ct = blockIdx.x & 15;
  int base = sym_off[b];
  int cnt = sym_off[b + 1] - base; if (cnt > MAXSYM) cnt = MAXSYM;
  int n0 = ct * 64;
  for (int i = t; i < H_; i += 256) { bPK[i] = prm[P_PRED_K_B + i]; bFK[i] = prm[P_ARG_FK_B + i]; }
  for (int i = t; i < 64 * H_; i += 256) {
    int r = i >> 7, c = i & 127, n = n0 + r;
    float x = (n < cnt) ? symbol_x[(size_t)(base + n) * H_ + c] : 0.f;
    Xl[r][c] = (_Float16)x;
  }
  __syncthreads();
#pragma unroll
  for (int j = 0; j < 8; ++j) {
    int tid = wv * 8 + j;
    int isF = tid >> 5, tl = tid & 31;
    int tm = tl >> 3, tn = tl & 7;
    const _Float16* wfp = isF ? wfFK : wfPK;
    v8f c = {};
#pragma unroll
    for (int tk = 0; tk < 4; ++tk) {
      v16h a = frag_ld(&Xl[0][0], H_, tm * 16, tk * 32, lane);
      v16h bm = wfrag_ld(wfp, tk, tn, lane);
      c = wmma16(a, bm, c);
    }
    const float* bias = isF ? bFK : bPK;
    _Float16 (*Out)[H_] = isF ? Fk : Kp;
    int col = tn * 16 + (lane & 15);
    int rb = tm * 16 + ((lane >> 4) << 3);
#pragma unroll
    for (int i = 0; i < 8; ++i) Out[rb + i][col] = (_Float16)(c[i] + bias[col]);
  }
  __syncthreads();
  const float scl = 0.08838834764831845f;  // 1/sqrt(128)
  if (wv < 4) {                             // pred logits: 4 column groups
    int g = wv;
    v8f c = {};
#pragma unroll
    for (int tk = 0; tk < 4; ++tk) {
      v16h a = frag_ld(predq_a + (size_t)b * 16 * H_, H_, 0, tk * 32, lane);
      v16h bm = frag_ld(&Kp[0][0], H_, g * 16, tk * 32, lane);
      c = wmma16(a, bm, c);
    }
    if (lane < 16) {
      int n = n0 + g * 16 + lane;
#pragma unroll
      for (int i = 0; i < KS; ++i)
        out[OFF_PRED + (size_t)(b * KS + i) * MAXSYM + n] = c[i] * scl;
    }
  }
  {                                         // arg_func logits: 2 q-tiles x 4 groups
    int at = wv >> 2, g = wv & 3;
    v8f c = {};
#pragma unroll
    for (int tk = 0; tk < 4; ++tk) {
      v16h a = frag_ld(afq_a + (size_t)b * 32 * H_, H_, at * 16, tk * 32, lane);
      v16h bm = frag_ld(&Fk[0][0], H_, g * 16, tk * 32, lane);
      c = wmma16(a, bm, c);
    }
    int n = n0 + g * 16 + (lane & 15);
#pragma unroll
    for (int i = 0; i < 8; ++i) {
      int q = at * 16 + i + ((lane >> 4) << 3);
      if (q < 24)
        out[OFF_AFUNC + (size_t)(b * 24 + q) * MAXSYM + n] = c[i] * scl;
    }
  }
}

// ---------------------------------------------------------------------------
// seq_to_slots: sequential scan, one thread per sample.
// ---------------------------------------------------------------------------
__global__ __launch_bounds__(64)
void k_seq2slots(const int* __restrict__ actions, const int* __restrict__ arguments,
                 const int* __restrict__ lengths, float* __restrict__ out) {
  int b = threadIdx.x;
  if (b >= BB) return;
  float activ[KS], pol[KS];
  int pred[KS], atype[KS][MA], avar[KS][MA], afunc[KS][MA];
  for (int i = 0; i < KS; ++i) {
    activ[i] = 0.f; pol[i] = 0.f; pred[i] = 0;
    for (int j = 0; j < MA; ++j) { atype[i][j] = 0; avar[i][j] = 0; afunc[i][j] = 0; }
  }
  int lit = -1, ai = 0, dp = 0;
  int L = lengths[b];
  for (int tt = 0; tt < TT; ++tt) {
    int act = actions[b * TT + tt], ag = arguments[b * TT + tt];
    bool valid = tt < L;
    bool is_new = valid && (act == 0 || act == 1);
    if (is_new) { lit += 1; ai = 0; dp = 0; }
    bool lit_ok = (lit >= 0) && (lit < KS);
    int sl = lit < 0 ? 0 : (lit >= KS ? KS - 1 : lit);
    if (is_new && lit_ok) { activ[sl] = 1.f; pol[sl] = (act == 1) ? 1.f : 0.f; }
    if (valid && act == 2 && lit_ok) pred[sl] = ag;
    int sa = ai < 0 ? 0 : (ai >= MA ? MA - 1 : ai);
    bool ok_arg = lit_ok && (ai < MA) && (dp == 0);
    bool is_var = valid && (act == 3) && ok_arg;
    if (is_var) { atype[sl][sa] = 1; avar[sl][sa] = (ag < 19) ? ag : 19; }
    bool is_func = valid && (act == 4) && ok_arg;
    if (is_func) { atype[sl][sa] = 2; afunc[sl][sa] = ag; }
    ai += is_var ? 1 : 0;
    dp += is_func ? 1 : 0;
    bool is_end = valid && (act == 5) && (dp > 0);
    dp -= is_end ? 1 : 0;
  }
  for (int i = 0; i < KS; ++i) {
    out[OFF_TACT + b * KS + i] = activ[i];
    out[OFF_TPOL + b * KS + i] = pol[i];
    out[OFF_TPRED + b * KS + i] = (float)pred[i];
    for (int j = 0; j < MA; ++j) {
      out[OFF_TATYPE + (b * KS + i) * MA + j] = (float)atype[i][j];
      out[OFF_TAVAR + (b * KS + i) * MA + j] = (float)avar[i][j];
      out[OFF_TAFUNC + (b * KS + i) * MA + j] = (float)afunc[i][j];
    }
  }
}

// ---------------------------------------------------------------------------
extern "C" void kernel_launch(void* const* d_in, const int* in_sizes, int n_in,
                              void* d_out, int out_size, void* d_ws, size_t ws_size,
                              hipStream_t stream) {
  (void)n_in; (void)out_size; (void)ws_size;
  const float* symbol_x = (const float*)d_in[0];
  const float* all_x    = (const float*)d_in[1];
  const float* prm      = (const float*)d_in[2];
  const int* sym_batch  = (const int*)d_in[3];
  const int* all_batch  = (const int*)d_in[4];
  const int* actions    = (const int*)d_in[5];
  const int* arguments  = (const int*)d_in[6];
  const int* lengths    = (const int*)d_in[7];
  int nsym = in_sizes[3];
  int nall = in_sizes[4];
  float* out = (float*)d_out;

  // workspace carve (deterministic, ~3.1 MB)
  char* w = (char*)d_ws;
  auto carve = [&](size_t bytes) { void* p = (void*)w; w += (bytes + 255) & ~(size_t)255; return p; };
  _Float16* wf      = (_Float16*)carve((size_t)6 * 16384 * sizeof(_Float16));
  _Float16* predq_a = (_Float16*)carve((size_t)BB * 16 * H_ * sizeof(_Float16));
  _Float16* afq_a   = (_Float16*)carve((size_t)BB * 32 * H_ * sizeof(_Float16));
  int* sym_off      = (int*)carve(65 * sizeof(int));
  int* all_off      = (int*)carve(65 * sizeof(int));
  float* slotsA     = (float*)carve((size_t)BB * KS * H_ * sizeof(float));
  float* slotsB     = (float*)carve((size_t)BB * KS * H_ * sizeof(float));
  float* o_part     = (float*)carve((size_t)BB * SPLIT * 24 * 32 * sizeof(float));
  float* m_part     = (float*)carve((size_t)BB * SPLIT * 24 * sizeof(float));
  float* l_part     = (float*)carve((size_t)BB * SPLIT * 24 * sizeof(float));

  int prep_items = 6 * 16384 + BB * KS * H_ + 130;
  k_prep<<<(prep_items + 255) / 256, 256, 0, stream>>>(prm, sym_batch, nsym, all_batch, nall,
                                                       sym_off, all_off, wf, slotsA);
  k_seq2slots<<<1, 64, 0, stream>>>(actions, arguments, lengths, out);

  // layer 1: ca1 + ln1 ; sa1 + ln2 + ffn1 + ln3
  k_cross_attn<<<BB * SPLIT, 256, 0, stream>>>(all_x, prm, all_off, slotsA,
                                               wf + 0 * 16384, wf + 1 * 16384, P_CA1,
                                               o_part, m_part, l_part);
  k_attn_combine<<<BB, 128, 0, stream>>>(prm, slotsA, o_part, m_part, l_part,
                                         P_CA1, P_LN1, slotsB);
  k_mid<<<BB, 128, 0, stream>>>(prm, slotsB, P_SA1, P_LN2, P_FFN1, P_LN3, slotsA);

  // layer 2: ca2 + ln4 ; sa2 + ln5 + ffn2 + ln6
  k_cross_attn<<<BB * SPLIT, 256, 0, stream>>>(all_x, prm, all_off, slotsA,
                                               wf + 2 * 16384, wf + 3 * 16384, P_CA2,
                                               o_part, m_part, l_part);
  k_attn_combine<<<BB, 128, 0, stream>>>(prm, slotsA, o_part, m_part, l_part,
                                         P_CA2, P_LN4, slotsB);
  k_mid<<<BB, 128, 0, stream>>>(prm, slotsB, P_SA2, P_LN5, P_FFN2, P_LN6, slotsA);

  // heads
  k_heads<<<BB, 128, 0, stream>>>(prm, slotsA, out, predq_a, afq_a);
  k_sym_logits<<<BB * 16, 256, 0, stream>>>(symbol_x, prm, sym_off,
                                            wf + 4 * 16384, wf + 5 * 16384,
                                            predq_a, afq_a, out);
}
